// LeNet_8744553415194
// MI455X (gfx1250) — matrix operand order repaired and checked by reading, not compiled, not run on hardware
//
#include <hip/hip_runtime.h>
#include <hip/hip_bf16.h>

typedef __attribute__((ext_vector_type(16))) __bf16 v16bf;
typedef __attribute__((ext_vector_type(8)))  float  v8f;

// ---------------------------------------------------------------------------
// Dims
// ---------------------------------------------------------------------------
#define BATCH   128
#define H1C     6
#define H1HW    112          // pooled conv1 output is [128,6,112,112]
#define H2C     16
#define H2HW    54           // pooled conv2 output is [128,16,54,54]
#define KFC1    46656        // 16*54*54
#define NFC1    1024
#define NFC2    128

// ---------------------------------------------------------------------------
// CDNA5 async global->LDS copy (ASYNCcnt-tracked, in-order completion).
// The LDS base pointer is passed as an operand so it ESCAPES into the asm:
// without it, alias analysis proves the asm cannot write the __shared__
// array and deletes every LDS read of data the async copy produced.
// ---------------------------------------------------------------------------
__device__ __forceinline__ void async_ld16(unsigned lds_byte_off, const void* gptr,
                                           void* lds_base) {
    asm volatile("global_load_async_to_lds_b128 %0, %1, off"
                 :: "v"(lds_byte_off), "v"(gptr), "v"(lds_base)
                 : "memory");
}

// ---------------------------------------------------------------------------
// Build dense [16,6,5,5] conv2 weight from the LeNet connection tables
// G1 row i = {i, i+1, i+2} mod 6 ; G2 row i = {i..i+3} mod 6 ;
// G3 row i = {i+0, i+1, i+3, i+4} mod 6
// ---------------------------------------------------------------------------
__global__ void build_w2(const float* __restrict__ w1, const float* __restrict__ b1,
                         const float* __restrict__ w2, const float* __restrict__ b2,
                         const float* __restrict__ w3, const float* __restrict__ b3,
                         const float* __restrict__ w4, const float* __restrict__ b4,
                         float* __restrict__ W, float* __restrict__ B) {
    int t = threadIdx.x;
    for (int i = t; i < 16 * 6 * 25; i += blockDim.x) W[i] = 0.f;
    __syncthreads();
    for (int i = t; i < 6 * 3 * 25; i += blockDim.x) {          // group 1
        int e = i % 25, j = (i / 25) % 3, br = i / 75;
        int ci = (br + j) % 6;
        W[(br * 6 + ci) * 25 + e] = w1[j * 25 + e];
    }
    for (int i = t; i < 6 * 4 * 25; i += blockDim.x) {          // group 2
        int e = i % 25, j = (i / 25) % 4, br = i / 100;
        int ci = (br + j) % 6;
        W[((6 + br) * 6 + ci) * 25 + e] = w2[j * 25 + e];
    }
    for (int i = t; i < 3 * 4 * 25; i += blockDim.x) {          // group 3
        int e = i % 25, j = (i / 25) % 4, br = i / 100;
        const int off[4] = {0, 1, 3, 4};
        int ci = (br + off[j]) % 6;
        W[((12 + br) * 6 + ci) * 25 + e] = w3[j * 25 + e];
    }
    for (int i = t; i < 6 * 25; i += blockDim.x) {              // group 4 (full)
        int e = i % 25, ci = i / 25;
        W[(15 * 6 + ci) * 25 + e] = w4[ci * 25 + e];
    }
    if (t < 16)
        B[t] = (t < 6) ? b1[0] : (t < 12) ? b2[0] : (t < 15) ? b3[0] : b4[0];
}

// ---------------------------------------------------------------------------
// conv1 (k3, pad2 on 222 -> 224) + ReLU + 2x2 maxpool -> [128,6,112,112] f32
// ---------------------------------------------------------------------------
__global__ void conv1_relu_pool(const float* __restrict__ x,
                                const float* __restrict__ w,
                                const float* __restrict__ bias,
                                float* __restrict__ out) {
    int idx = blockIdx.x * blockDim.x + threadIdx.x;
    const int total = BATCH * H1C * H1HW * H1HW;
    if (idx >= total) return;
    int px = idx % H1HW; int t = idx / H1HW;
    int py = t % H1HW;   t /= H1HW;
    int c  = t % H1C;    int b = t / H1C;

    const float* xb = x + (size_t)b * 3 * 222 * 222;
    const float* wc = w + c * 27;
    float bb = bias[c];
    float best = 0.f;                       // relu then max == max(0, max s)
    #pragma unroll
    for (int dy = 0; dy < 2; ++dy)
    #pragma unroll
    for (int dx = 0; dx < 2; ++dx) {
        int oy = 2 * py + dy, ox = 2 * px + dx;   // conv coords in 224x224
        float s = bb;
        for (int ci = 0; ci < 3; ++ci) {
            const float* xc = xb + (size_t)ci * 222 * 222;
            const float* wk = wc + ci * 9;
            #pragma unroll
            for (int ky = 0; ky < 3; ++ky) {
                int iy = oy - 2 + ky;
                if ((unsigned)iy < 222u) {
                    #pragma unroll
                    for (int kx = 0; kx < 3; ++kx) {
                        int ix = ox - 2 + kx;
                        if ((unsigned)ix < 222u)
                            s = fmaf(xc[iy * 222 + ix], wk[ky * 3 + kx], s);
                    }
                }
            }
        }
        best = fmaxf(best, s);
    }
    out[idx] = best;
}

// ---------------------------------------------------------------------------
// dense conv2 (16,6,5,5) pad0 on 112 -> 108, +ReLU + 2x2 pool -> 54,
// written as bf16 flattened GEMM-A rows [128, 46656]
// ---------------------------------------------------------------------------
__global__ void conv2_relu_pool(const float* __restrict__ h,
                                const float* __restrict__ W,
                                const float* __restrict__ B,
                                __hip_bfloat16* __restrict__ out) {
    int idx = blockIdx.x * blockDim.x + threadIdx.x;
    const int total = BATCH * H2C * H2HW * H2HW;
    if (idx >= total) return;
    int px = idx % H2HW; int t = idx / H2HW;
    int py = t % H2HW;   t /= H2HW;
    int c  = t % H2C;    int b = t / H2C;

    float bb = B[c];
    float best = 0.f;
    #pragma unroll
    for (int dy = 0; dy < 2; ++dy)
    #pragma unroll
    for (int dx = 0; dx < 2; ++dx) {
        int oy = 2 * py + dy, ox = 2 * px + dx;   // in [0,107]
        float s = bb;
        for (int ci = 0; ci < 6; ++ci) {
            const float* hc = h + ((size_t)(b * 6 + ci)) * (112 * 112);
            const float* wk = W + (c * 6 + ci) * 25;
            #pragma unroll
            for (int ky = 0; ky < 5; ++ky) {
                const float* row = hc + (oy + ky) * 112 + ox;
                #pragma unroll
                for (int kx = 0; kx < 5; ++kx)
                    s = fmaf(row[kx], wk[ky * 5 + kx], s);
            }
        }
        best = fmaxf(best, s);
    }
    out[idx] = __float2bfloat16(best);
}

// ---------------------------------------------------------------------------
// fp32 -> bf16 conversion (grid-stride)
// ---------------------------------------------------------------------------
__global__ void cvt_bf16(const float* __restrict__ in,
                         __hip_bfloat16* __restrict__ out, long n) {
    long i = (long)blockIdx.x * blockDim.x + threadIdx.x;
    long stride = (long)gridDim.x * blockDim.x;
    for (; i < n; i += stride) out[i] = __float2bfloat16(in[i]);
}

// ---------------------------------------------------------------------------
// bf16 WMMA GEMM:  C[M,N] = A[M,K] * Bw[N,K]^T   (both K-major, fp32 accum)
// 64x64 macro tile / block (256 threads = 8 waves: 4 along M, 2 along N),
// each wave produces a 16x32 slab via two v_wmma_f32_16x16x32_bf16 accums.
// K staged through LDS in 32-wide steps using async global->LDS copies,
// double buffered on ASYNCcnt (in-order completion): issue tile t+1 into
// the alternate buffer, s_wait_asynccnt 2 to drain tile t, barrier, compute.
// ---------------------------------------------------------------------------
__global__ void __launch_bounds__(256)
gemm_bf16_wmma(const __hip_bfloat16* __restrict__ A,
               const __hip_bfloat16* __restrict__ Bw,
               float* __restrict__ C, int M, int N, int K) {
    // [buf(2)] x { A: 64x32 bf16 (4KB) | B: 64x32 bf16 (4KB) }  -> 16KB
    __shared__ __hip_bfloat16 smem[2 * 4096];

    const int tid   = threadIdx.x;
    const int wave  = tid >> 5;
    const int lane  = tid & 31;
    const int mwave = wave & 3;           // 0..3 -> 16-row slabs
    const int nwave = wave >> 2;          // 0..1 -> 32-col slabs
    const int m0 = blockIdx.x * 64;
    const int n0 = blockIdx.y * 64;
    const int half = lane >> 4;           // lane half per ISA layout
    const int lm   = lane & 15;

    // cooperative staging: 256 threads x 16B per matrix = 4KB per tile
    const int lrow = tid >> 2;            // 0..63
    const int lcol = (tid & 3) * 8;       // 0,8,16,24 (bf16 elems)

    const unsigned ldsA = (unsigned)(lrow * 32 + lcol) * 2;   // bytes (buf 0)
    const unsigned ldsB = ldsA + 4096;
    const size_t aoff = (size_t)(m0 + lrow) * K + lcol;
    const size_t boff = (size_t)(n0 + lrow) * K + lcol;

    v8f acc0 = {};
    v8f acc1 = {};

    // prologue: tile 0 -> buffer 0
    async_ld16(ldsA, A + aoff, smem);
    async_ld16(ldsB, Bw + boff, smem);

    int cur = 0;
    for (int k0 = 0; k0 < K; k0 += 32) {
        if (k0 + 32 < K) {
            unsigned nb = (unsigned)(cur ^ 1) * 8192u;   // bytes
            async_ld16(ldsA + nb, A + aoff + k0 + 32, smem);
            async_ld16(ldsB + nb, Bw + boff + k0 + 32, smem);
            // 2 copies of the NEXT tile stay in flight; current tile landed
            asm volatile("s_wait_asynccnt 2" ::: "memory");
        } else {
            asm volatile("s_wait_asynccnt 0" ::: "memory");
        }
        __syncthreads();

        const __hip_bfloat16* As = smem + cur * 4096;
        const __hip_bfloat16* Bs = As + 2048;

        // A fragment: lane holds row (mwave*16+lm); half 0 -> K 0..15,
        // half 1 -> K 16..31 (same contiguous K order as B)
        const int arow = mwave * 16 + lm;
        union { uint4 q[2]; v16bf v; } af, bf0, bf1;
        af.q[0] = *(const uint4*)(&As[arow * 32 + half * 16]);
        af.q[1] = *(const uint4*)(&As[arow * 32 + half * 16 + 8]);

        // B fragment: lane holds column n (lm), half selects K 0..15 / 16..31
        const int bcol0 = nwave * 32 + lm;
        bf0.q[0] = *(const uint4*)(&Bs[bcol0 * 32 + half * 16]);
        bf0.q[1] = *(const uint4*)(&Bs[bcol0 * 32 + half * 16 + 8]);
        const int bcol1 = bcol0 + 16;
        bf1.q[0] = *(const uint4*)(&Bs[bcol1 * 32 + half * 16]);
        bf1.q[1] = *(const uint4*)(&Bs[bcol1 * 32 + half * 16 + 8]);

        acc0 = __builtin_amdgcn_wmma_f32_16x16x32_bf16(
                   false, af.v, false, bf0.v, (short)0, acc0, false, false);
        acc1 = __builtin_amdgcn_wmma_f32_16x16x32_bf16(
                   false, af.v, false, bf1.v, (short)0, acc1, false, false);

        __syncthreads();   // all reads of `cur` done before it is re-targeted
        cur ^= 1;
    }

    // C/D layout: lane -> col n=lm, VGPR r -> row r + 8*half
    const int crow = m0 + mwave * 16 + half * 8;
    const int ccol = n0 + nwave * 32 + lm;
    #pragma unroll
    for (int r = 0; r < 8; ++r) {
        C[(size_t)(crow + r) * N + ccol]      = acc0[r];
        C[(size_t)(crow + r) * N + ccol + 16] = acc1[r];
    }
}

// ---------------------------------------------------------------------------
// train-mode BatchNorm over batch (M=128) + ReLU.  One block per column.
// (Pre-BN bias is omitted upstream: mean subtraction cancels it exactly.)
// ---------------------------------------------------------------------------
__global__ void bn_relu(const float* __restrict__ Y,
                        const float* __restrict__ g,
                        const float* __restrict__ be,
                        __hip_bfloat16* __restrict__ out_bf,
                        float* __restrict__ out_f, int N) {
    const int col = blockIdx.x;
    const int m   = threadIdx.x;          // blockDim.x == 128 == BATCH
    float v = Y[(size_t)m * N + col];
    __shared__ float ssum[128], ssq[128];
    ssum[m] = v; ssq[m] = v * v;
    __syncthreads();
    #pragma unroll
    for (int s = 64; s > 0; s >>= 1) {
        if (m < s) { ssum[m] += ssum[m + s]; ssq[m] += ssq[m + s]; }
        __syncthreads();
    }
    float mean = ssum[0] * (1.0f / 128.0f);
    float var  = ssq[0] * (1.0f / 128.0f) - mean * mean;   // biased, as x.var(0)
    float o = fmaxf(g[col] * (v - mean) * rsqrtf(var + 1e-5f) + be[col], 0.f);
    if (out_bf) out_bf[(size_t)m * N + col] = __float2bfloat16(o);
    if (out_f)  out_f [(size_t)m * N + col] = o;
}

// ---------------------------------------------------------------------------
// fc3: [128,128] @ [10,128]^T + b -> [128,10]
// ---------------------------------------------------------------------------
__global__ void fc3_kernel(const float* __restrict__ H,
                           const float* __restrict__ W,
                           const float* __restrict__ b,
                           float* __restrict__ out) {
    int idx = blockIdx.x * blockDim.x + threadIdx.x;
    if (idx >= BATCH * 10) return;
    int m = idx / 10, j = idx % 10;
    const float* h = H + m * 128;
    const float* w = W + j * 128;
    float s = b[j];
    #pragma unroll 4
    for (int k = 0; k < 128; ++k) s = fmaf(h[k], w[k], s);
    out[idx] = s;
}

// ---------------------------------------------------------------------------
static inline size_t align_up(size_t v, size_t a) { return (v + a - 1) & ~(a - 1); }

extern "C" void kernel_launch(void* const* d_in, const int* in_sizes, int n_in,
                              void* d_out, int out_size, void* d_ws, size_t ws_size,
                              hipStream_t stream) {
    const float* x     = (const float*)d_in[0];
    const float* w_c1  = (const float*)d_in[1];
    const float* b_c1  = (const float*)d_in[2];
    const float* w1    = (const float*)d_in[3];
    const float* b1    = (const float*)d_in[4];
    const float* w2    = (const float*)d_in[5];
    const float* b2    = (const float*)d_in[6];
    const float* w3    = (const float*)d_in[7];
    const float* b3    = (const float*)d_in[8];
    const float* w4    = (const float*)d_in[9];
    const float* b4    = (const float*)d_in[10];
    const float* w_fc1 = (const float*)d_in[11];
    /* b_fc1 (d_in[12]) cancelled by BN mean subtraction */
    const float* g_bn1 = (const float*)d_in[13];
    const float* be_bn1= (const float*)d_in[14];
    const float* w_fc2 = (const float*)d_in[15];
    /* b_fc2 (d_in[16]) cancelled by BN mean subtraction */
    const float* g_bn2 = (const float*)d_in[17];
    const float* be_bn2= (const float*)d_in[18];
    const float* w_fc3 = (const float*)d_in[19];
    const float* b_fc3 = (const float*)d_in[20];

    // ---- workspace carve-out -------------------------------------------
    char* p = (char*)d_ws;
    size_t off = 0;
    auto carve = [&](size_t bytes) -> void* {
        void* r = p + off;
        off = align_up(off + bytes, 256);
        return r;
    };
    float*          h1       = (float*)carve((size_t)BATCH * H1C * H1HW * H1HW * 4);
    float*          W2d      = (float*)carve(16 * 6 * 25 * 4);
    float*          B2d      = (float*)carve(16 * 4);
    __hip_bfloat16* act1     = (__hip_bfloat16*)carve((size_t)BATCH * KFC1 * 2);
    __hip_bfloat16* wfc1_bf  = (__hip_bfloat16*)carve((size_t)NFC1 * KFC1 * 2);
    float*          y1       = (float*)carve((size_t)BATCH * NFC1 * 4);
    __hip_bfloat16* hbn1     = (__hip_bfloat16*)carve((size_t)BATCH * NFC1 * 2);
    __hip_bfloat16* wfc2_bf  = (__hip_bfloat16*)carve((size_t)NFC2 * NFC1 * 2);
    float*          y2       = (float*)carve((size_t)BATCH * NFC2 * 4);
    float*          hbn2     = (float*)carve((size_t)BATCH * NFC2 * 4);
    (void)ws_size;

    // 1) dense conv2 weight
    build_w2<<<1, 256, 0, stream>>>(w1, b1, w2, b2, w3, b3, w4, b4, W2d, B2d);

    // 2) conv1 + relu + pool
    {
        int total = BATCH * H1C * H1HW * H1HW;
        conv1_relu_pool<<<(total + 255) / 256, 256, 0, stream>>>(x, w_c1, b_c1, h1);
    }

    // 3) conv2 + relu + pool -> bf16 flattened activations
    {
        int total = BATCH * H2C * H2HW * H2HW;
        conv2_relu_pool<<<(total + 255) / 256, 256, 0, stream>>>(h1, W2d, B2d, act1);
    }

    // 4) weight conversions to bf16
    cvt_bf16<<<8192, 256, 0, stream>>>(w_fc1, wfc1_bf, (long)NFC1 * KFC1);
    cvt_bf16<<<512,  256, 0, stream>>>(w_fc2, wfc2_bf, (long)NFC2 * NFC1);

    // 5) fc1 GEMM: [128,46656] x [1024,46656]^T -> [128,1024]
    {
        dim3 grid(BATCH / 64, NFC1 / 64);
        gemm_bf16_wmma<<<grid, 256, 0, stream>>>(act1, wfc1_bf, y1,
                                                 BATCH, NFC1, KFC1);
    }

    // 6) BN1 + relu -> bf16
    bn_relu<<<NFC1, 128, 0, stream>>>(y1, g_bn1, be_bn1, hbn1, nullptr, NFC1);

    // 7) fc2 GEMM: [128,1024] x [128,1024]^T -> [128,128]
    {
        dim3 grid(BATCH / 64, NFC2 / 64);
        gemm_bf16_wmma<<<grid, 256, 0, stream>>>(hbn1, wfc2_bf, y2,
                                                 BATCH, NFC2, NFC1);
    }

    // 8) BN2 + relu -> f32
    bn_relu<<<NFC2, 128, 0, stream>>>(y2, g_bn2, be_bn2, nullptr, hbn2, NFC2);

    // 9) fc3 -> d_out [128,10]
    fc3_kernel<<<(BATCH * 10 + 63) / 64, 64, 0, stream>>>(hbn2, w_fc3, b_fc3,
                                                          (float*)d_out);
}
